// TransformerEncoder_3685081940268
// MI455X (gfx1250) — matrix operand order repaired and checked by reading, not compiled, not run on hardware
//
#include <hip/hip_runtime.h>

#define B_   4
#define L_   8192
#define D_   256
#define H_   8
#define DH_  32
#define M_   128
#define BL_  (B_*L_)

#define DN_    0.4204482076268573f   // 32^-0.25
#define RATIO_ 0.08838834764831845f  // 1/sqrt(128)
#define STAB_  1e-6f
#define EPSLN_ 1e-3f

typedef __bf16 bf16;
typedef __attribute__((ext_vector_type(16))) __bf16 v16bf;
typedef __attribute__((ext_vector_type(4)))  __bf16 v4bf;
typedef __attribute__((ext_vector_type(8)))  float  v8f;

__device__ __forceinline__ bf16 f2bf(float x){ return (bf16)x; }

__device__ __forceinline__ unsigned lds_off(const void* p){
  return (unsigned)(uintptr_t)p;            // addrspace(3) ptr -> LDS byte offset
}

// async copy 16B global->LDS, tracked by ASYNCcnt (gfx1250)
__device__ __forceinline__ void async_b128(unsigned ldsAddr, unsigned long long gAddr){
  asm volatile("global_load_async_to_lds_b128 %0, %1, off"
               :: "v"(ldsAddr), "v"(gAddr) : "memory");
}

// A fragment 16x32 bf16 (ISA 7.12.2): lanes 0-15 row M=lane, K pairs 0..7 & 16..23;
// lanes 16-31 same rows, K pairs 8..15 & 24..31.  src row-major, stride lda.
__device__ __forceinline__ v16bf load_a(const bf16* s, int lda, int mBase, int kBase, int lane){
  int m  = mBase + (lane & 15);
  int kh = (lane >> 4) << 3;
  v16bf a;
#pragma unroll
  for (int v = 0; v < 8; ++v){
    int k = kBase + ((v < 4) ? (v*2 + kh) : (16 + (v-4)*2 + kh));
    a[2*v]   = s[m*lda + k];
    a[2*v+1] = s[m*lda + k + 1];
  }
  return a;
}

// B fragment 32x16 bf16: lane 0-15 col N=lane with K 0..7 & 16..23; lanes 16-31 same cols K 8..15 & 24..31.
// src stored N-major ([N][K], stride ldk) so pairs are contiguous.
__device__ __forceinline__ v16bf load_b(const bf16* s, int ldk, int nBase, int kBase, int lane){
  int n  = nBase + (lane & 15);
  int kh = (lane >> 4) << 3;
  v16bf b;
#pragma unroll
  for (int v = 0; v < 8; ++v){
    int k = kBase + ((v < 4) ? (v*2 + kh) : (16 + (v-4)*2 + kh));
    b[2*v]   = s[n*ldk + k];
    b[2*v+1] = s[n*ldk + k + 1];
  }
  return b;
}

__global__ __launch_bounds__(256) void cvt_bf16(const float* __restrict__ src, bf16* __restrict__ dst,
                                                int rows, int cols, int transpose){
  int i = blockIdx.x*256 + threadIdx.x;
  if (i >= rows*cols) return;
  int r = i / cols, c = i % cols;
  dst[transpose ? (c*rows + r) : i] = f2bf(src[i]);
}

// C[16 x 256] = A[16 x 256] @ B[256 x 256] (+bias)(+residual)(+relu)(+layernorm)
// BwT is bf16 N-major: BwT[n*256 + k].   flags: 1=relu, 2=residual, 4=layernorm
// B panels (32 K x 256 N = 16KB) stream to LDS via double-buffered async copies.
__global__ __launch_bounds__(256) void gemm_fused(
    const float* __restrict__ A, const bf16* __restrict__ BwT,
    const float* __restrict__ bias, const float* __restrict__ resid,
    const float* __restrict__ gamma, const float* __restrict__ beta,
    float* __restrict__ out, int flags)
{
  __shared__ alignas(16) bf16  lB[2][256*32];  // N-major [n][k], 2 x 16KB
  __shared__ alignas(16) bf16  lA[2][16*32];
  __shared__ float lO[16*256];
  const int tid = threadIdx.x, lane = tid & 31, w = tid >> 5;
  const int row0 = blockIdx.x * 16;

  // issue async B panel kk into buffer buf: row tid, 64B = 4 x B128
  auto issueB = [&](int kk, int buf){
    unsigned long long g = (unsigned long long)(uintptr_t)(BwT + (size_t)tid*256 + kk*32);
    unsigned l = lds_off(&lB[buf][tid*32]);
#pragma unroll
    for (int j = 0; j < 4; ++j) async_b128(l + j*16, g + j*16);
  };
  // stage A tile (16x32 f32 -> bf16) into lA[buf]
  auto loadA = [&](int kk, int buf){
    if (tid < 128){
      int r = tid >> 3, c4 = (tid & 7) * 4;
      float4 f = *(const float4*)&A[(size_t)(row0 + r)*256 + kk*32 + c4];
      v4bf bv; bv[0] = f2bf(f.x); bv[1] = f2bf(f.y); bv[2] = f2bf(f.z); bv[3] = f2bf(f.w);
      *(v4bf*)&lA[buf][r*32 + c4] = bv;
    }
  };

  v8f acc0 = {}; v8f acc1 = {};
  issueB(0, 0);
  for (int kk = 0; kk < 8; ++kk){
    int buf = kk & 1;
    loadA(kk, buf);
    if (kk < 7) issueB(kk + 1, buf ^ 1);
    // retire current panel's 4 async ops (in-order completion), keep next panel in flight
    if (kk < 7) asm volatile("s_wait_asynccnt 0x4" ::: "memory");
    else        asm volatile("s_wait_asynccnt 0x0" ::: "memory");
    __syncthreads();
    v16bf a  = load_a(lA[buf], 32, 0, 0, lane);
    v16bf b0 = load_b(lB[buf], 32, w*32,      0, lane);
    v16bf b1 = load_b(lB[buf], 32, w*32 + 16, 0, lane);
    acc0 = __builtin_amdgcn_wmma_f32_16x16x32_bf16(false, a, false, b0, (short)0, acc0, false, false);
    acc1 = __builtin_amdgcn_wmma_f32_16x16x32_bf16(false, a, false, b1, (short)0, acc1, false, false);
  }
  __syncthreads();
  {
    int n = lane & 15, mh = (lane >> 4) * 8;
#pragma unroll
    for (int r = 0; r < 8; ++r){
      int m = r + mh;
      lO[m*256 + w*32 + n]      = acc0[r];
      lO[m*256 + w*32 + 16 + n] = acc1[r];
    }
  }
  __syncthreads();
  for (int i = tid; i < 16*256; i += 256){
    int m = i >> 8, c = i & 255;
    float v = lO[i] + bias[c];
    if (flags & 2) v += resid[(size_t)(row0 + m)*256 + c];
    if (flags & 1) v = fmaxf(v, 0.f);
    if (flags & 4) lO[i] = v;
    else out[(size_t)(row0 + m)*256 + c] = v;
  }
  if (flags & 4){
    __syncthreads();
#pragma unroll
    for (int rr = 0; rr < 2; ++rr){
      int m = w*2 + rr;
      float s = 0.f, s2 = 0.f;
#pragma unroll
      for (int j = 0; j < 8; ++j){
        float v = lO[m*256 + lane + j*32];
        s += v; s2 += v*v;
      }
#pragma unroll
      for (int off = 16; off > 0; off >>= 1){
        s  += __shfl_xor(s,  off, 32);
        s2 += __shfl_xor(s2, off, 32);
      }
      float mu  = s  * (1.f/256.f);
      float var = s2 * (1.f/256.f) - mu*mu;
      float rs  = rsqrtf(var + EPSLN_);
#pragma unroll
      for (int j = 0; j < 8; ++j){
        int c = lane + j*32;
        out[(size_t)(row0 + m)*256 + c] = (lO[m*256 + c] - mu) * rs * gamma[c] + beta[c];
      }
    }
  }
}

// diag[bl*H+h] = 0.5 * sum((x*dn)^2) over head slice
__global__ __launch_bounds__(256) void diag_kernel(const float* __restrict__ X, float* __restrict__ diag){
  int idx = blockIdx.x*256 + threadIdx.x;
  if (idx >= BL_*H_) return;
  int bl = idx >> 3, h = idx & 7;
  const float4* p = (const float4*)(X + (size_t)bl*256 + h*32);
  float s = 0.f;
#pragma unroll
  for (int j = 0; j < 8; ++j){
    float4 v = p[j];
    s += v.x*v.x + v.y*v.y + v.z*v.z + v.w*v.w;
  }
  diag[idx] = 0.5f * DN_ * DN_ * s;
}

// dash[64 rows x 128 feats] = (X_head * dn) @ proj^T  (K=32, one WMMA step).
// projB is bf16 [M=128][DH=32] row-major == N-major B, async-copied whole (8KB).
__global__ __launch_bounds__(256) void phi_gemm(const float* __restrict__ X, const bf16* __restrict__ projB,
                                                float* __restrict__ dash, float* __restrict__ blockmax, int doMax)
{
  __shared__ alignas(16) bf16  lA[64*32];
  __shared__ alignas(16) bf16  lB[128*32];   // N-major
  __shared__ float lred[256];
  const int tid = threadIdx.x, lane = tid & 31, w = tid >> 5;
  const int row0 = blockIdx.x * 64;
  const int h = blockIdx.y;
  {
    // async: 8KB, 32B per thread = 2 x B128
    unsigned long long g = (unsigned long long)(uintptr_t)projB + (unsigned)tid*32u;
    unsigned l = lds_off(&lB[0]) + tid*32;
    async_b128(l,      g);
    async_b128(l + 16, g + 16);
  }
  for (int i = tid; i < 512; i += 256){           // 64x32 floats as float4
    int r = i >> 3, c4 = (i & 7) * 4;
    float4 f = *(const float4*)&X[(size_t)(row0 + r)*256 + h*32 + c4];
    v4bf bv; bv[0] = f2bf(f.x*DN_); bv[1] = f2bf(f.y*DN_); bv[2] = f2bf(f.z*DN_); bv[3] = f2bf(f.w*DN_);
    *(v4bf*)&lA[r*32 + c4] = bv;
  }
  asm volatile("s_wait_asynccnt 0x0" ::: "memory");
  __syncthreads();
  const int wr = w >> 1, wc = w & 1;
  v16bf a = load_a(lA, 32, wr*16, 0, lane);
  v8f acc[4];
#pragma unroll
  for (int t = 0; t < 4; ++t){
    v8f c0 = {};
    v16bf b = load_b(lB, 32, wc*64 + t*16, 0, lane);
    acc[t] = __builtin_amdgcn_wmma_f32_16x16x32_bf16(false, a, false, b, (short)0, c0, false, false);
  }
  float mx = -3.0e38f;
  int n = lane & 15, mh = (lane >> 4)*8;
#pragma unroll
  for (int t = 0; t < 4; ++t){
#pragma unroll
    for (int r = 0; r < 8; ++r){
      int m   = wr*16 + r + mh;
      int col = wc*64 + t*16 + n;
      float v = acc[t][r];
      dash[(size_t)(row0 + m)*1024 + h*128 + col] = v;
      mx = fmaxf(mx, v);
    }
  }
  if (doMax){
    lred[tid] = mx; __syncthreads();
    for (int s = 128; s > 0; s >>= 1){
      if (tid < s) lred[tid] = fmaxf(lred[tid], lred[tid + s]);
      __syncthreads();
    }
    if (tid == 0) blockmax[blockIdx.y * gridDim.x + blockIdx.x] = lred[0];
  }
}

__global__ __launch_bounds__(256) void reduce_max_kernel(const float* __restrict__ part, int n, float* __restrict__ gmax){
  __shared__ float lred[256];
  float mx = -3.0e38f;
  for (int i = threadIdx.x; i < n; i += 256) mx = fmaxf(mx, part[i]);
  lred[threadIdx.x] = mx; __syncthreads();
  for (int s = 128; s > 0; s >>= 1){
    if (threadIdx.x < s) lred[threadIdx.x] = fmaxf(lred[threadIdx.x], lred[threadIdx.x+s]);
    __syncthreads();
  }
  if (threadIdx.x == 0) *gmax = lred[0];
}

// in place: dash = ratio*(exp(dash - diag - max) + stab); queries per-row max, keys global
__global__ __launch_bounds__(256) void phi_exp(float* __restrict__ dash, const float* __restrict__ diag,
                                               const float* __restrict__ gmax, int useGlobal)
{
  int row = blockIdx.x*256 + threadIdx.x;
  if (row >= BL_*H_) return;
  float4* p = (float4*)(dash + (size_t)row*128);
  float mx;
  if (useGlobal) mx = *gmax;
  else {
    mx = -3.0e38f;
    for (int i = 0; i < 32; ++i){
      float4 v = p[i];
      mx = fmaxf(mx, fmaxf(fmaxf(v.x, v.y), fmaxf(v.z, v.w)));
    }
  }
  float dg = diag[row] + mx;
  for (int i = 0; i < 32; ++i){
    float4 v = p[i];
    v.x = RATIO_ * (__expf(v.x - dg) + STAB_);
    v.y = RATIO_ * (__expf(v.y - dg) + STAB_);
    v.z = RATIO_ * (__expf(v.z - dg) + STAB_);
    v.w = RATIO_ * (__expf(v.w - dg) + STAB_);
    p[i] = v;
  }
}

// kv[b,h,m,dh] = sum_l kp[b,l,h,m]*v[b,l,h,dh] ; ksum[b,h,m] = sum_l kp
__global__ __launch_bounds__(256) void kv_kernel(const float* __restrict__ kp, const float* __restrict__ V,
                                                 float* __restrict__ kv, float* __restrict__ ksum)
{
  __shared__ float lkp[128];
  __shared__ float lv[32];
  const int tid = threadIdx.x;
  const int b = blockIdx.x >> 3, h = blockIdx.x & 7;
  const int dh = tid & 31, mb = (tid >> 5) * 16;
  float acc[16];
#pragma unroll
  for (int j = 0; j < 16; ++j) acc[j] = 0.f;
  float ssum = 0.f;
  for (int l = 0; l < L_; ++l){
    size_t bl = (size_t)b*L_ + l;
    if (tid < 128)      lkp[tid]      = kp[bl*1024 + h*128 + tid];
    else if (tid < 160) lv[tid - 128] = V[bl*256 + h*32 + (tid - 128)];
    __syncthreads();
    float vv = lv[dh];
#pragma unroll
    for (int j = 0; j < 16; ++j) acc[j] = fmaf(lkp[mb + j], vv, acc[j]);
    if (tid < 128) ssum += lkp[tid];
    __syncthreads();
  }
#pragma unroll
  for (int j = 0; j < 16; ++j)
    kv[((size_t)blockIdx.x*128 + mb + j)*32 + dh] = acc[j];
  if (tid < 128) ksum[(size_t)blockIdx.x*128 + tid] = ssum;
}

// attn[64 rows x 32 dh] = (qp[64x128] @ kv[128x32]) / (qp @ ksum)   (K=128, 4 WMMA steps)
__global__ __launch_bounds__(256) void attn_kernel(const float* __restrict__ qp, const float* __restrict__ kvM,
                                                   const float* __restrict__ ksum, float* __restrict__ attn)
{
  __shared__ alignas(16) bf16  lA[64*128];
  __shared__ alignas(16) bf16  lB[32*128];   // N-major [dh][k]
  __shared__ float lks[128];
  __shared__ float lpart[256];
  __shared__ float lden[64];
  const int tid = threadIdx.x, lane = tid & 31, w = tid >> 5;
  const int row0 = blockIdx.x * 64;
  const int h = blockIdx.y;
  const int b = row0 >> 13;
  const float* kvp = kvM  + (size_t)(b*H_ + h)*128*32;
  const float* ksp = ksum + (size_t)(b*H_ + h)*128;

  for (int i = tid; i < 2048; i += 256){          // 64x128 floats as float4
    int r = i >> 5, c4 = (i & 31) * 4;
    float4 f = *(const float4*)&qp[(size_t)(row0 + r)*1024 + h*128 + c4];
    v4bf bv; bv[0] = f2bf(f.x); bv[1] = f2bf(f.y); bv[2] = f2bf(f.z); bv[3] = f2bf(f.w);
    *(v4bf*)&lA[r*128 + c4] = bv;
  }
  for (int i = tid; i < 32*128; i += 256){
    int n = i >> 7, k = i & 127;
    lB[i] = f2bf(kvp[(size_t)k*32 + n]);
  }
  if (tid < 128) lks[tid] = ksp[tid];
  __syncthreads();

  const int wr = w >> 1, wc = w & 1;
  v8f acc = {};
#pragma unroll
  for (int kk = 0; kk < 4; ++kk){
    v16bf a  = load_a(lA, 128, wr*16, kk*32, lane);
    v16bf bb = load_b(lB, 128, wc*16, kk*32, lane);
    acc = __builtin_amdgcn_wmma_f32_16x16x32_bf16(false, a, false, bb, (short)0, acc, false, false);
  }
  {
    int r = tid >> 2, q = tid & 3;
    float s = 0.f;
    for (int k = q*32; k < q*32 + 32; ++k) s += (float)lA[r*128 + k] * lks[k];
    lpart[tid] = s;
  }
  __syncthreads();
  if (tid < 64) lden[tid] = lpart[tid*4] + lpart[tid*4+1] + lpart[tid*4+2] + lpart[tid*4+3];
  __syncthreads();
  {
    int n = lane & 15, mh = (lane >> 4)*8;
#pragma unroll
    for (int r = 0; r < 8; ++r){
      int m = wr*16 + r + mh;
      attn[(size_t)(row0 + m)*256 + h*32 + wc*16 + n] = acc[r] / lden[m];
    }
  }
}

extern "C" void kernel_launch(void* const* d_in, const int* in_sizes, int n_in,
                              void* d_out, int out_size, void* d_ws, size_t ws_size,
                              hipStream_t stream)
{
  (void)in_sizes; (void)n_in; (void)out_size; (void)ws_size;
  const float* x     = (const float*)d_in[0];
  const float* wq    = (const float*)d_in[1];
  const float* bq    = (const float*)d_in[2];
  const float* wk    = (const float*)d_in[3];
  const float* bk    = (const float*)d_in[4];
  const float* wv    = (const float*)d_in[5];
  const float* bv    = (const float*)d_in[6];
  const float* wo    = (const float*)d_in[7];
  const float* bo    = (const float*)d_in[8];
  const float* proj  = (const float*)d_in[9];
  const float* g0    = (const float*)d_in[10];
  const float* beta0 = (const float*)d_in[11];
  const float* w0    = (const float*)d_in[12];
  const float* bfc0  = (const float*)d_in[13];
  const float* w1    = (const float*)d_in[14];
  const float* bfc1  = (const float*)d_in[15];
  const float* g1    = (const float*)d_in[16];
  const float* beta1 = (const float*)d_in[17];
  float* out = (float*)d_out;

  char* ws = (char*)d_ws;
  size_t off = 0;
  auto take = [&](size_t bytes)->char*{ char* p = ws + off; off += (bytes + 255) & ~(size_t)255; return p; };
  float* q     = (float*)take((size_t)BL_*256*4);   // later aliased: attn
  float* k     = (float*)take((size_t)BL_*256*4);   // later aliased: xn
  float* v     = (float*)take((size_t)BL_*256*4);   // later aliased: h0
  float* qd    = (float*)take((size_t)BL_*1024*4);  // qdash -> qp (in place)
  float* kd    = (float*)take((size_t)BL_*1024*4);  // kdash -> kp (in place)
  float* diagQ = (float*)take((size_t)BL_*H_*4);
  float* diagK = (float*)take((size_t)BL_*H_*4);
  float* maxp  = (float*)take(4096*4);
  float* gmax  = (float*)take(256);
  float* kvb   = (float*)take((size_t)B_*H_*128*32*4);
  float* ksb   = (float*)take((size_t)B_*H_*128*4);
  bf16* bwq = (bf16*)take(65536*2);   // all N-major bf16
  bf16* bwk = (bf16*)take(65536*2);
  bf16* bwv = (bf16*)take(65536*2);
  bf16* bwo = (bf16*)take(65536*2);
  bf16* bw0 = (bf16*)take(65536*2);
  bf16* bw1 = (bf16*)take(65536*2);
  bf16* bpj = (bf16*)take(4096*2);    // proj [128][32] bf16 (already N-major)
  float* attn = q;
  float* xn   = k;
  float* h0   = v;

  dim3 blk(256);
  cvt_bf16<<<256, blk, 0, stream>>>(wq, bwq, 256, 256, 1);  // -> [n][k]
  cvt_bf16<<<256, blk, 0, stream>>>(wk, bwk, 256, 256, 1);
  cvt_bf16<<<256, blk, 0, stream>>>(wv, bwv, 256, 256, 1);
  cvt_bf16<<<256, blk, 0, stream>>>(wo, bwo, 256, 256, 1);
  cvt_bf16<<<256, blk, 0, stream>>>(w0, bw0, 256, 256, 1);
  cvt_bf16<<<256, blk, 0, stream>>>(w1, bw1, 256, 256, 1);
  cvt_bf16<<<16,  blk, 0, stream>>>(proj, bpj, 128, 32, 0);

  gemm_fused<<<BL_/16, blk, 0, stream>>>(x, bwq, bq, x, bq, bq, q, 0);
  gemm_fused<<<BL_/16, blk, 0, stream>>>(x, bwk, bk, x, bk, bk, k, 0);
  gemm_fused<<<BL_/16, blk, 0, stream>>>(x, bwv, bv, x, bv, bv, v, 0);

  diag_kernel<<<BL_*H_/256, blk, 0, stream>>>(q, diagQ);
  diag_kernel<<<BL_*H_/256, blk, 0, stream>>>(k, diagK);

  phi_gemm<<<dim3(BL_/64, H_), blk, 0, stream>>>(q, bpj, qd, maxp, 0);
  phi_gemm<<<dim3(BL_/64, H_), blk, 0, stream>>>(k, bpj, kd, maxp, 1);
  reduce_max_kernel<<<1, blk, 0, stream>>>(maxp, (BL_/64)*H_, gmax);
  phi_exp<<<BL_*H_/256, blk, 0, stream>>>(qd, diagQ, gmax, 0);
  phi_exp<<<BL_*H_/256, blk, 0, stream>>>(kd, diagK, gmax, 1);

  kv_kernel<<<B_*H_, blk, 0, stream>>>(kd, v, kvb, ksb);
  attn_kernel<<<dim3(BL_/64, H_), blk, 0, stream>>>(qd, kvb, ksb, attn);

  gemm_fused<<<BL_/16, blk, 0, stream>>>(attn, bwo, bo,  x, g0, beta0, xn,  2|4);  // +res +LN
  gemm_fused<<<BL_/16, blk, 0, stream>>>(xn,  bw0, bfc0, x, g0, beta0, h0,  1);    // relu
  gemm_fused<<<BL_/16, blk, 0, stream>>>(h0,  bw1, bfc1, x, g1, beta1, out, 1|4);  // relu + LN
}
